// AttentionOperation_20358144983086
// MI455X (gfx1250) — matrix operand order repaired and checked by reading, not compiled, not run on hardware
//
#include <hip/hip_runtime.h>
#include <math.h>

// Problem constants (from reference setup_inputs)
#define NN 8
#define HH 8
#define DD 64
#define LL 16384
#define MM 256
#define CC 64
#define BN_EPS 1e-3

typedef __bf16 v16bf __attribute__((ext_vector_type(16)));
typedef float v8f __attribute__((ext_vector_type(8)));
typedef unsigned short v16us __attribute__((ext_vector_type(16)));
typedef unsigned short v8us __attribute__((ext_vector_type(8)));

// ---------- bf16 split helpers (bit-exact, no __bf16 arithmetic needed) ----------
__device__ __forceinline__ unsigned short f32_to_bf16_rne(float x) {
    unsigned u = __builtin_bit_cast(unsigned, x);
    unsigned r = u + 0x7FFFu + ((u >> 16) & 1u);
    return (unsigned short)(r >> 16);
}
__device__ __forceinline__ float bf16_to_f32(unsigned short h) {
    return __builtin_bit_cast(float, ((unsigned)h) << 16);
}
__device__ __forceinline__ void splitf(float x, unsigned short& h, unsigned short& l) {
    h = f32_to_bf16_rne(x);
    l = f32_to_bf16_rne(x - bf16_to_f32(h));
}

__device__ __forceinline__ v8f wmma_bf16(v16bf a, v16bf b, v8f c) {
    // D = A(16x32) x B(32x16) + C, f32 accumulate
    return __builtin_amdgcn_wmma_f32_16x16x32_bf16(false, a, false, b, (short)0, c, false, false);
}

__device__ __forceinline__ float gelu_exact(float y) {
    return 0.5f * y * (1.0f + erff(y * 0.70710678118654752440f));
}

// ---------- K0: zero workspace accumulators ----------
__global__ void zero_kernel(float* __restrict__ p, int n) {
    int i = blockIdx.x * 256 + threadIdx.x;
    if (i < n) p[i] = 0.0f;
}

// ---------- K0a: pack K f32[bh][d][m] -> transposed bf16 planes [bh][m][d] ----------
__global__ void pack_kt_kernel(const float* __restrict__ X,
                               unsigned short* __restrict__ Hi,
                               unsigned short* __restrict__ Lo, int n) {
    int i = blockIdx.x * 256 + threadIdx.x;
    if (i < n) {
        int m = i % MM;
        int d = (i / MM) % DD;
        int bh = i / (MM * DD);
        unsigned short h, l;
        splitf(X[i], h, l);
        size_t o = ((size_t)bh * MM + m) * DD + d;
        Hi[o] = h; Lo[o] = l;
    }
}

// ---------- K0b: pack V f32 -> bf16 planes, same [bh][c][m] layout ----------
__global__ void pack_v_kernel(const float* __restrict__ X,
                              unsigned short* __restrict__ Hi,
                              unsigned short* __restrict__ Lo, int n) {
    int i = blockIdx.x * 256 + threadIdx.x;
    if (i < n) {
        unsigned short h, l;
        splitf(X[i], h, l);
        Hi[i] = h; Lo[i] = l;
    }
}

// ---------- K1/K2: Gram matrix G = X X^T (64x64) per (b,h), plus row-sums ----------
// BN2d stats:  sum(sim^2) = <G_Q, G_K>_F,  sum(sim) = rowsum_Q . rowsum_K
__global__ __launch_bounds__(256) void gram_kernel(const float* __restrict__ X,
                                                   float* __restrict__ G,
                                                   float* __restrict__ RS,
                                                   int len, int chunks) {
    __shared__ float xs[64][65];
    int bh = blockIdx.x / chunks;
    int ck = blockIdx.x % chunks;
    int t = threadIdx.x;
    const float* Xb = X + (size_t)bh * 64 * len;
    int clen = len / chunks;
    int l0 = ck * clen;
    int steps = clen / 64;

    float acc[16];
#pragma unroll
    for (int i = 0; i < 16; ++i) acc[i] = 0.0f;
    float rs = 0.0f;
    int pd[16], pd2[16];
#pragma unroll
    for (int i = 0; i < 16; ++i) { int p = t + 256 * i; pd[i] = p >> 6; pd2[i] = p & 63; }

    for (int s = 0; s < steps; ++s) {
#pragma unroll
        for (int j = 0; j < 16; ++j) {
            int i = t + 256 * j;
            int d = i >> 6, ll = i & 63;
            xs[d][ll] = Xb[(size_t)d * len + l0 + s * 64 + ll];
        }
        __syncthreads();
        for (int ll = 0; ll < 64; ++ll) {
#pragma unroll
            for (int i = 0; i < 16; ++i) acc[i] += xs[pd[i]][ll] * xs[pd2[i]][ll];
        }
        if (t < 64) {
            float r2 = 0.0f;
            for (int ll = 0; ll < 64; ++ll) r2 += xs[t][ll];
            rs += r2;
        }
        __syncthreads();
    }
#pragma unroll
    for (int i = 0; i < 16; ++i) atomicAdd(&G[(size_t)bh * 4096 + t + 256 * i], acc[i]);
    if (t < 64) atomicAdd(&RS[(size_t)bh * 64 + t], rs);
}

// ---------- K3: finalize BN2d scale/bias per h ----------
__global__ __launch_bounds__(256) void bn2_fin_kernel(const float* __restrict__ GQ,
                                                      const float* __restrict__ GK,
                                                      const float* __restrict__ RSQ,
                                                      const float* __restrict__ RSK,
                                                      const float* __restrict__ gamma,
                                                      const float* __restrict__ beta,
                                                      float* __restrict__ SC2,
                                                      float* __restrict__ BI2) {
    __shared__ double red[256];
    int t = threadIdx.x;
    for (int h = 0; h < HH; ++h) {
        double ssq = 0.0;
        for (int i = t; i < NN * 4096; i += 256) {
            int b = i >> 12, j = i & 4095;
            size_t o = (size_t)(b * HH + h) * 4096 + j;
            ssq += (double)GQ[o] * (double)GK[o];
        }
        double sm = 0.0;
        for (int i = t; i < NN * 64; i += 256) {
            int b = i >> 6, d = i & 63;
            size_t o = (size_t)(b * HH + h) * 64 + d;
            sm += (double)RSQ[o] * (double)RSK[o];
        }
        red[t] = ssq; __syncthreads();
        for (int s = 128; s > 0; s >>= 1) { if (t < s) red[t] += red[t + s]; __syncthreads(); }
        double tssq = red[0]; __syncthreads();
        red[t] = sm; __syncthreads();
        for (int s = 128; s > 0; s >>= 1) { if (t < s) red[t] += red[t + s]; __syncthreads(); }
        double tsm = red[0]; __syncthreads();
        if (t == 0) {
            const double cnt = (double)NN * (double)LL * (double)MM;
            double mean = tsm / cnt;
            double var = tssq / cnt - mean * mean;
            float sc = gamma[h] * (float)(1.0 / sqrt(var + (double)BN_EPS));
            SC2[h] = sc;
            BI2[h] = beta[h] - (float)mean * sc;
        }
        __syncthreads();
    }
}

// ---------- K4: fused sim -> BN2d affine -> softmax -> attn*V^T -> store rv + BN1d sums ----------
// Grid: (N*H) * (L/64) blocks, 128 threads = 4 waves, each wave owns 16 L-rows.
__global__ __launch_bounds__(128) void attn_rv_kernel(const float* __restrict__ Q,
                                                      const unsigned short* __restrict__ KhiT,
                                                      const unsigned short* __restrict__ KloT,
                                                      const unsigned short* __restrict__ Vhi,
                                                      const unsigned short* __restrict__ Vlo,
                                                      const float* __restrict__ SC2,
                                                      const float* __restrict__ BI2,
                                                      double* __restrict__ S1,
                                                      double* __restrict__ SQ1,
                                                      float* __restrict__ Out) {
    const int ltiles = LL / 64;
    int bh = blockIdx.x / ltiles;
    int lt = blockIdx.x % ltiles;
    int b = bh >> 3;
    int h = bh & 7;
    int tid = threadIdx.x;
    int wave = tid >> 5;
    int lane = tid & 31;
    int half = lane >> 4;
    int r = lane & 15;

    extern __shared__ unsigned short slds[];           // 64KB total
    unsigned short* Ahi = slds + (size_t)wave * 2 * 16 * MM;   // 16x256 bf16 hi plane
    unsigned short* Alo = Ahi + 16 * MM;                       // 16x256 bf16 lo plane

    const float* Qb = Q + (size_t)bh * DD * LL;
    const unsigned short* KhiB = KhiT + (size_t)bh * MM * DD;  // [m][d]
    const unsigned short* KloB = KloT + (size_t)bh * MM * DD;
    const unsigned short* VhiB = Vhi + (size_t)bh * CC * MM;   // [c][m]
    const unsigned short* VloB = Vlo + (size_t)bh * CC * MM;

    float s2 = SC2[h], bi2 = BI2[h];
    int lbase = lt * 64 + wave * 16;

    // ---- sim = Q^T K for 16 rows, 256 cols : 16 WMMA C tiles ----
    v8f acc[16];
#pragma unroll
    for (int i = 0; i < 16; ++i)
#pragma unroll
        for (int j = 0; j < 8; ++j) acc[i][j] = 0.0f;

#pragma unroll
    for (int kt = 0; kt < 2; ++kt) {
        // A operand: A[l][k] = Q[d=kt*32+k][lbase+l]  (bf16 hi/lo split in-register)
        v16us ahu, alu;
#pragma unroll
        for (int e = 0; e < 16; ++e) {
            int k = half * 16 + e;
            int d = kt * 32 + k;
            unsigned short hh, ll2;
            splitf(Qb[(size_t)d * LL + lbase + r], hh, ll2);
            ahu[e] = hh; alu[e] = ll2;
        }
        v16bf ah = __builtin_bit_cast(v16bf, ahu);
        v16bf al = __builtin_bit_cast(v16bf, alu);
#pragma unroll
        for (int mt = 0; mt < 16; ++mt) {
            // B operand: B[k][n] = K[d][mt*16+n]; transposed planes -> 32B contiguous per lane
            size_t ko = (size_t)(mt * 16 + r) * DD + kt * 32 + half * 16;
            v16us bhu = *(const v16us*)(KhiB + ko);
            v16us blu = *(const v16us*)(KloB + ko);
            v16bf bh16 = __builtin_bit_cast(v16bf, bhu);
            v16bf bl16 = __builtin_bit_cast(v16bf, blu);
            acc[mt] = wmma_bf16(ah, bh16, acc[mt]);   // hi*hi
            acc[mt] = wmma_bf16(ah, bl16, acc[mt]);   // hi*lo
            acc[mt] = wmma_bf16(al, bh16, acc[mt]);   // lo*hi
        }
    }

    // ---- BN2d affine + softmax over m (row = half*8+v lives on one 16-lane group) ----
    float mx[8];
#pragma unroll
    for (int v = 0; v < 8; ++v) mx[v] = -3.0e38f;
#pragma unroll
    for (int i = 0; i < 16; ++i)
#pragma unroll
        for (int v = 0; v < 8; ++v) {
            float x = acc[i][v] * s2 + bi2;
            acc[i][v] = x;
            mx[v] = fmaxf(mx[v], x);
        }
#pragma unroll
    for (int v = 0; v < 8; ++v) {
        mx[v] = fmaxf(mx[v], __shfl_xor(mx[v], 1, 32));
        mx[v] = fmaxf(mx[v], __shfl_xor(mx[v], 2, 32));
        mx[v] = fmaxf(mx[v], __shfl_xor(mx[v], 4, 32));
        mx[v] = fmaxf(mx[v], __shfl_xor(mx[v], 8, 32));
    }
    float sme[8];
#pragma unroll
    for (int v = 0; v < 8; ++v) sme[v] = 0.0f;
#pragma unroll
    for (int i = 0; i < 16; ++i)
#pragma unroll
        for (int v = 0; v < 8; ++v) {
            float e = __expf(acc[i][v] - mx[v]);
            acc[i][v] = e;
            sme[v] += e;
        }
#pragma unroll
    for (int v = 0; v < 8; ++v) {
        float s = sme[v];
        s += __shfl_xor(s, 1, 32);
        s += __shfl_xor(s, 2, 32);
        s += __shfl_xor(s, 4, 32);
        s += __shfl_xor(s, 8, 32);
        sme[v] = 1.0f / s;
    }

    // ---- C-layout -> LDS as pre-split bf16 hi/lo planes (split once at store) ----
#pragma unroll
    for (int i = 0; i < 16; ++i)
#pragma unroll
        for (int v = 0; v < 8; ++v) {
            float a = acc[i][v] * sme[v];
            unsigned short hh, ll2;
            splitf(a, hh, ll2);
            int idx = (half * 8 + v) * MM + i * 16 + r;
            Ahi[idx] = hh;
            Alo[idx] = ll2;
        }
    __syncthreads();

    // ---- rv[l][c] = sum_m attn[l][m] * V[c][m] : 4 WMMA C tiles, K=256 ----
    v8f acc2[4];
#pragma unroll
    for (int i = 0; i < 4; ++i)
#pragma unroll
        for (int j = 0; j < 8; ++j) acc2[i][j] = 0.0f;

    for (int kt = 0; kt < 8; ++kt) {
        // A operand from LDS planes: two 16B contiguous runs per plane
        int ao = r * MM + kt * 32 + half * 16;
        v8us h0 = *(const v8us*)(Ahi + ao);
        v8us h1 = *(const v8us*)(Ahi + ao + 8);
        v8us l0 = *(const v8us*)(Alo + ao);
        v8us l1 = *(const v8us*)(Alo + ao + 8);
        v16us ahu, alu;
#pragma unroll
        for (int e = 0; e < 8; ++e) {
            ahu[e] = h0[e]; ahu[8 + e] = h1[e];
            alu[e] = l0[e]; alu[8 + e] = l1[e];
        }
        v16bf ah = __builtin_bit_cast(v16bf, ahu);
        v16bf al = __builtin_bit_cast(v16bf, alu);
#pragma unroll
        for (int ct = 0; ct < 4; ++ct) {
            // B operand: V planes, 32B contiguous per lane
            size_t vo = (size_t)(ct * 16 + r) * MM + kt * 32 + half * 16;
            v16us bhu = *(const v16us*)(VhiB + vo);
            v16us blu = *(const v16us*)(VloB + vo);
            v16bf bh16 = __builtin_bit_cast(v16bf, bhu);
            v16bf bl16 = __builtin_bit_cast(v16bf, blu);
            acc2[ct] = wmma_bf16(ah, bh16, acc2[ct]);
            acc2[ct] = wmma_bf16(ah, bl16, acc2[ct]);
            acc2[ct] = wmma_bf16(al, bh16, acc2[ct]);
        }
    }
    __syncthreads();

    // ---- stage rv [c=64][l=64] in LDS for coalesced stores + BN1d partial sums ----
    float* rv_lds = (float*)slds;
#pragma unroll
    for (int ct = 0; ct < 4; ++ct)
#pragma unroll
        for (int v = 0; v < 8; ++v)
            rv_lds[(ct * 16 + r) * 64 + wave * 16 + half * 8 + v] = acc2[ct][v];
    __syncthreads();

    int c = tid >> 1;
    int loff = (tid & 1) * 32;
    const float* src = rv_lds + c * 64 + loff;
    size_t obase = ((size_t)b * (HH * CC) + (size_t)h * CC + c) * LL + (size_t)lt * 64 + loff;
    double ds = 0.0, dss = 0.0;
#pragma unroll
    for (int i = 0; i < 32; i += 4) {
        float4 x = *(const float4*)(src + i);
        *(float4*)(Out + obase + i) = x;
        ds += (double)x.x + (double)x.y + (double)x.z + (double)x.w;
        dss += (double)x.x * x.x + (double)x.y * x.y + (double)x.z * x.z + (double)x.w * x.w;
    }
    atomicAdd(&S1[h * CC + c], ds);
    atomicAdd(&SQ1[h * CC + c], dss);
}

// ---------- K5: finalize BN1d scale/bias per channel ----------
__global__ void bn1_fin_kernel(const double* __restrict__ S1, const double* __restrict__ SQ1,
                               const float* __restrict__ g1, const float* __restrict__ b1,
                               float* __restrict__ SC1, float* __restrict__ BI1) {
    int ch = blockIdx.x * 256 + threadIdx.x;
    if (ch < HH * CC) {
        const double cnt = (double)NN * (double)LL;
        double mean = S1[ch] / cnt;
        double var = SQ1[ch] / cnt - mean * mean;
        float sc = g1[ch] * (float)(1.0 / sqrt(var + (double)BN_EPS));
        SC1[ch] = sc;
        BI1[ch] = b1[ch] - (float)mean * sc;
    }
}

// ---------- K6: BN1d affine + exact GELU, in-place over d_out ----------
__global__ __launch_bounds__(256) void bn_gelu_kernel(float* __restrict__ Out,
                                                      const float* __restrict__ SC1,
                                                      const float* __restrict__ BI1) {
    size_t i4 = (size_t)blockIdx.x * 256 + threadIdx.x;
    size_t base = i4 * 4;
    int ch = (int)((base / LL) % (HH * CC));
    float sc = SC1[ch], bi = BI1[ch];
    float4 x = *(const float4*)(Out + base);
    x.x = gelu_exact(x.x * sc + bi);
    x.y = gelu_exact(x.y * sc + bi);
    x.z = gelu_exact(x.z * sc + bi);
    x.w = gelu_exact(x.w * sc + bi);
    *(float4*)(Out + base) = x;
}

extern "C" void kernel_launch(void* const* d_in, const int* in_sizes, int n_in,
                              void* d_out, int out_size, void* d_ws, size_t ws_size,
                              hipStream_t stream) {
    (void)in_sizes; (void)n_in; (void)out_size; (void)ws_size;
    const float* Qf = (const float*)d_in[0];
    const float* Kf = (const float*)d_in[1];
    const float* Vf = (const float*)d_in[2];
    const float* g2 = (const float*)d_in[3];
    const float* b2 = (const float*)d_in[4];
    const float* g1 = (const float*)d_in[5];
    const float* b1 = (const float*)d_in[6];
    float* Out = (float*)d_out;
    float* ws = (float*)d_ws;

    // workspace layout (float offsets)
    float* GQ = ws;                         // 262144
    float* GK = ws + 262144;                // 262144
    float* RSQ = ws + 524288;               // 4096
    float* RSK = ws + 528384;               // 4096
    float* SC2 = ws + 532480;               // 8
    float* BI2 = ws + 532488;               // 8
    float* SC1 = ws + 532496;               // 512
    float* BI1 = ws + 533008;               // 512
    double* S1 = (double*)(ws + 533520);    // 512 doubles
    double* SQ1 = (double*)(ws + 534544);   // 512 doubles
    // bf16 planes (each 1048576 u16 = 524288 floats worth)
    unsigned short* KhiT = (unsigned short*)(ws + 540672);
    unsigned short* KloT = (unsigned short*)(ws + 1064960);
    unsigned short* VhiP = (unsigned short*)(ws + 1589248);
    unsigned short* VloP = (unsigned short*)(ws + 2113536);

    const int nk = NN * HH * DD * MM;   // 1048576
    const int nv = NN * HH * CC * MM;   // 1048576

    zero_kernel<<<(535568 + 255) / 256, 256, 0, stream>>>(ws, 535568);
    pack_kt_kernel<<<(nk + 255) / 256, 256, 0, stream>>>(Kf, KhiT, KloT, nk);
    pack_v_kernel<<<(nv + 255) / 256, 256, 0, stream>>>(Vf, VhiP, VloP, nv);
    gram_kernel<<<NN * HH * 16, 256, 0, stream>>>(Qf, GQ, RSQ, LL, 16);
    gram_kernel<<<NN * HH, 256, 0, stream>>>(Kf, GK, RSK, MM, 1);
    bn2_fin_kernel<<<1, 256, 0, stream>>>(GQ, GK, RSQ, RSK, g2, b2, SC2, BI2);
    attn_rv_kernel<<<NN * HH * (LL / 64), 128, 65536, stream>>>(Qf, KhiT, KloT, VhiP, VloP,
                                                                SC2, BI2, S1, SQ1, Out);
    bn1_fin_kernel<<<2, 256, 0, stream>>>(S1, SQ1, g1, b1, SC1, BI1);
    bn_gelu_kernel<<<(NN * HH * CC * LL) / (4 * 256), 256, 0, stream>>>(Out, SC1, BI1);
}
